// LegendreKANLayer_7249904796068
// MI455X (gfx1250) — matrix operand order repaired and checked by reading, not compiled, not run on hardware
//
#include <hip/hip_runtime.h>

// ---------------------------------------------------------------------------
// LegendreKAN layer for gfx1250 (MI455X), wave32 + WMMA bf16.
//   BATCH=8192, INPUT_DIM=256, OUTPUT_DIM=256, NUM_CENTERS=512, ORDER=8
// Pipeline:
//   1) init min/max scalars
//   2) convert x/centers -> bf16, weights -> bf16 transposed [f][k] (k=c*9+n)
//   3) row norms x2, c2
//   4) distance GEMM (wmma bf16, 16x64 strip/wave) + sqrt + global min/max
//   5) fused Legendre-basis + big GEMM (wmma bf16, 16x64 strip/wave) -> out
// Each wave carries 4 independent accumulators so back-to-back WMMAs have no
// RAW hazard on D, and the A-fragment is reused across 4 B-tiles.
// ---------------------------------------------------------------------------

#define BATCH   8192
#define IN_DIM  256
#define OUT_DIM 256
#define NC      512
#define NORD    9              // order + 1
#define KTOT    (NC * NORD)    // 4608, divisible by 32
#define GC      32             // centers per LDS chunk
#define KC      (GC * NORD)    // 288 = 9 * 32
#define MROWS   32             // batch rows per block in main kernel
#define NTJ     4              // 16-wide N tiles per wave (N strip = 64)

typedef __attribute__((ext_vector_type(16))) __bf16 v16bf;
typedef __attribute__((ext_vector_type(8)))  float  v8f;

union Frag {
    v16bf v;
    uint4 q[2];
};

static __device__ __forceinline__ unsigned short f2bf(float f) {
    unsigned int u = __float_as_uint(f);
    unsigned int r = 0x7FFFu + ((u >> 16) & 1u);   // round-to-nearest-even
    return (unsigned short)((u + r) >> 16);
}

// ---------------------------------------------------------------- init ------
__global__ void init_minmax(unsigned int* mm) {
    if (threadIdx.x == 0) {
        mm[0] = 0x7F800000u;   // +inf bits (min accumulator)
        mm[1] = 0u;            // 0.0 bits  (max accumulator, dist >= 0)
    }
}

// ------------------------------------------------------------- converts -----
__global__ void cvt_bf16(const float* __restrict__ src,
                         unsigned short* __restrict__ dst, int n) {
    int i = blockIdx.x * blockDim.x + threadIdx.x;
    if (i < n) dst[i] = f2bf(src[i]);
}

// weights [512,9,256] row-major (= [k=c*9+n][f]) -> wBT[f][k] bf16
__global__ void cvt_weights_T(const float* __restrict__ w,
                              unsigned short* __restrict__ wBT) {
    int i = blockIdx.x * blockDim.x + threadIdx.x;   // i = k*OUT_DIM + f
    if (i < KTOT * OUT_DIM) {
        int f = i & (OUT_DIM - 1);
        int k = i >> 8;                               // OUT_DIM == 256
        wBT[(size_t)f * KTOT + k] = f2bf(w[i]);
    }
}

// ------------------------------------------------------------ row norms -----
__global__ void row_norms(const float* __restrict__ src,
                          float* __restrict__ dst, int rows, int cols) {
    int gwave = (blockIdx.x * blockDim.x + threadIdx.x) >> 5;
    int lane  = threadIdx.x & 31;
    if (gwave >= rows) return;
    const float* p = src + (size_t)gwave * cols;
    float s = 0.0f;
    for (int k = lane; k < cols; k += 32) { float v = p[k]; s += v * v; }
    for (int o = 16; o; o >>= 1) s += __shfl_xor(s, o, 32);
    if (lane == 0) dst[gwave] = s;
}

// ----------------------------------------------------- distances + minmax ---
// grid (BATCH/16, NC/(64*4)), block 128 (4 waves); each wave: 16x64 strip.
__global__ __launch_bounds__(128) void dist_kernel(
    const unsigned short* __restrict__ xBF,   // [BATCH][IN_DIM] bf16
    const unsigned short* __restrict__ cBF,   // [NC][IN_DIM] bf16
    const float* __restrict__ x2,
    const float* __restrict__ c2,
    float* __restrict__ dist,                 // [BATCH][NC]
    unsigned int* __restrict__ mm) {
    const int lane    = threadIdx.x & 31;
    const int wave    = threadIdx.x >> 5;
    const int rowBase = blockIdx.x * 16;
    const int colBase = (blockIdx.y * 4 + wave) * 64;

    const int halfA = (lane < 16) ? 0 : 16;   // A: contiguous 16-half group
    const int kB0   = (lane < 16) ? 0 : 16;   // B: contiguous 16-half group
    const unsigned short* xrow = xBF + (size_t)(rowBase + (lane & 15)) * IN_DIM;
    const unsigned short* crow[NTJ];
#pragma unroll
    for (int j = 0; j < NTJ; ++j)
        crow[j] = cBF + (size_t)(colBase + j * 16 + (lane & 15)) * IN_DIM;

    v8f acc[NTJ] = {};
#pragma unroll
    for (int k0 = 0; k0 < IN_DIM; k0 += 32) {
        Frag a;
        a.q[0] = *(const uint4*)(xrow + k0 + halfA);
        a.q[1] = *(const uint4*)(xrow + k0 + halfA + 8);
#pragma unroll
        for (int j = 0; j < NTJ; ++j) {
            Frag b;
            b.q[0] = *(const uint4*)(crow[j] + k0 + kB0);
            b.q[1] = *(const uint4*)(crow[j] + k0 + kB0 + 8);
            acc[j] = __builtin_amdgcn_wmma_f32_16x16x32_bf16(
                false, a.v, false, b.v, (short)0, acc[j], false, false);
        }
    }

    const int row0 = rowBase + ((lane < 16) ? 0 : 8);
    float xv[8];
#pragma unroll
    for (int r = 0; r < 8; ++r) xv[r] = x2[row0 + r];

    float vmin = 3.4e38f, vmax = 0.0f;
#pragma unroll
    for (int j = 0; j < NTJ; ++j) {
        const int col  = colBase + j * 16 + (lane & 15);
        const float c2v = c2[col];
#pragma unroll
        for (int r = 0; r < 8; ++r) {
            float d2 = xv[r] + c2v - 2.0f * acc[j][r];
            float d  = sqrtf(fmaxf(d2, 0.0f));
            dist[(size_t)(row0 + r) * NC + col] = d;
            vmin = fminf(vmin, d);
            vmax = fmaxf(vmax, d);
        }
    }
    for (int o = 16; o; o >>= 1) {
        vmin = fminf(vmin, __shfl_xor(vmin, o, 32));
        vmax = fmaxf(vmax, __shfl_xor(vmax, o, 32));
    }
    if (lane == 0) {   // dist >= 0 -> uint ordering == float ordering
        atomicMin(&mm[0], __float_as_uint(vmin));
        atomicMax(&mm[1], __float_as_uint(vmax));
    }
}

// --------------------------------------- fused basis + contraction GEMM -----
// grid (BATCH/MROWS, OUT_DIM/(16*NTJ)), block 64 (2 waves).
// Each wave: 16 rows x 64 cols (4 accumulators); basis tile shared in LDS.
__global__ __launch_bounds__(64) void kan_main(
    const float* __restrict__ dist,            // [BATCH][NC]
    const unsigned short* __restrict__ wBT,    // [OUT_DIM][KTOT] bf16
    const unsigned int* __restrict__ mm,
    float* __restrict__ out) {                 // [BATCH][OUT_DIM]
    __shared__ __align__(16) unsigned short sB[MROWS * KC];   // 18 KB

    const int tid     = threadIdx.x;
    const int lane    = tid & 31;
    const int wave    = tid >> 5;
    const int rowBase = blockIdx.x * MROWS;
    const int colBase = blockIdx.y * (16 * NTJ);

    const float dmin = __uint_as_float(mm[0]);
    const float dmax = __uint_as_float(mm[1]);
    const float inv  = 2.0f / (dmax - dmin);

    const int halfA = (lane < 16) ? 0 : 16;
    const int kB0   = (lane < 16) ? 0 : 16;
    const int rowA  = (wave << 4) + (lane & 15);             // 0..31
    const unsigned short* wcol[NTJ];
#pragma unroll
    for (int j = 0; j < NTJ; ++j)
        wcol[j] = wBT + (size_t)(colBase + j * 16 + (lane & 15)) * KTOT;

    v8f acc[NTJ] = {};
    for (int cc = 0; cc < NC / GC; ++cc) {
        const int cBase = cc * GC;
        // ---- stage Legendre basis tile [MROWS][GC*NORD] into LDS ----
        for (int p = tid; p < MROWS * GC; p += 64) {
            int r  = p >> 5;
            int cl = p & 31;
            float d = dist[(size_t)(rowBase + r) * NC + cBase + cl];
            float t = (d - dmin) * inv - 1.0f;
            unsigned short* dstp = &sB[r * KC + cl * NORD];
            float pm1 = 1.0f, pn = t;
            dstp[0] = f2bf(1.0f);
            dstp[1] = f2bf(t);
#pragma unroll
            for (int n = 1; n < NORD - 1; ++n) {
                float px = ((2.0f * n + 1.0f) * t * pn - (float)n * pm1) *
                           (1.0f / (float)(n + 1));
                pm1 = pn;
                pn  = px;
                dstp[n + 1] = f2bf(px);
            }
        }
        __syncthreads();
        // ---- 9 WMMA K-steps over this chunk (KC = 9*32), 4 N-tiles ----
#pragma unroll
        for (int ks = 0; ks < KC / 32; ++ks) {
            const int k0 = ks * 32;
            Frag a;
            a.q[0] = *(const uint4*)&sB[rowA * KC + k0 + halfA];
            a.q[1] = *(const uint4*)&sB[rowA * KC + k0 + halfA + 8];
#pragma unroll
            for (int j = 0; j < NTJ; ++j) {
                Frag b;
                const unsigned short* bp =
                    wcol[j] + (size_t)cBase * NORD + k0 + kB0;
                b.q[0] = *(const uint4*)bp;
                b.q[1] = *(const uint4*)(bp + 8);
                acc[j] = __builtin_amdgcn_wmma_f32_16x16x32_bf16(
                    false, a.v, false, b.v, (short)0, acc[j], false, false);
            }
        }
        __syncthreads();
    }

    const int outRow = rowBase + (wave << 4) + ((lane < 16) ? 0 : 8);
#pragma unroll
    for (int j = 0; j < NTJ; ++j) {
        const int outCol = colBase + j * 16 + (lane & 15);
#pragma unroll
        for (int r = 0; r < 8; ++r)
            out[(size_t)(outRow + r) * OUT_DIM + outCol] = acc[j][r];
    }
}

// ---------------------------------------------------------------- launch ----
extern "C" void kernel_launch(void* const* d_in, const int* in_sizes, int n_in,
                              void* d_out, int out_size, void* d_ws,
                              size_t ws_size, hipStream_t stream) {
    (void)in_sizes; (void)n_in; (void)out_size; (void)ws_size;

    const float* x       = (const float*)d_in[0];   // [8192,256]
    const float* centers = (const float*)d_in[1];   // [512,256]
    const float* weights = (const float*)d_in[2];   // [512,9,256]
    float* out = (float*)d_out;                     // [8192,256]

    char* ws = (char*)d_ws;
    // workspace layout (256-byte aligned slabs), total ~23.6 MB
    unsigned int*   mm   = (unsigned int*)(ws + 0);
    float*          x2   = (float*)(ws + 256);
    float*          c2   = (float*)(ws + 33024);
    float*          dist = (float*)(ws + 35072);                  // 16 MB
    unsigned short* xBF  = (unsigned short*)(ws + 16812288);      // 4 MB
    unsigned short* cBF  = (unsigned short*)(ws + 21006592);      // 256 KB
    unsigned short* wBT  = (unsigned short*)(ws + 21268736);      // 2.25 MB

    init_minmax<<<1, 32, 0, stream>>>(mm);

    cvt_bf16<<<(BATCH * IN_DIM) / 256, 256, 0, stream>>>(x, xBF,
                                                         BATCH * IN_DIM);
    cvt_bf16<<<(NC * IN_DIM) / 256, 256, 0, stream>>>(centers, cBF,
                                                      NC * IN_DIM);
    cvt_weights_T<<<(KTOT * OUT_DIM) / 256, 256, 0, stream>>>(weights, wBT);

    row_norms<<<(BATCH * 32) / 256, 256, 0, stream>>>(x, x2, BATCH, IN_DIM);
    row_norms<<<(NC * 32) / 256, 256, 0, stream>>>(centers, c2, NC, IN_DIM);

    dist_kernel<<<dim3(BATCH / 16, NC / 256), 128, 0, stream>>>(
        xBF, cBF, x2, c2, dist, mm);

    kan_main<<<dim3(BATCH / MROWS, OUT_DIM / (16 * NTJ)), 64, 0, stream>>>(
        dist, wBT, mm, out);
}
